// FIFLayer_84619445665820
// MI455X (gfx1250) — matrix operand order repaired and checked by reading, not compiled, not run on hardware
//
#include <hip/hip_runtime.h>
#include <hip/hip_bf16.h>

typedef __attribute__((ext_vector_type(16))) __bf16 v16bf;
typedef __attribute__((ext_vector_type(8)))  float  v8f;

#define ETA 0.15f
#define LAM 0.05f
#define KITER 6

#define BSZ   32
#define LSEQ  8192
#define DDIM  64
#define HID   256
#define TL    112
#define HALO  6
#define ROWS  (TL + 2*HALO)   // 124 valid rows
#define NT_L  ((LSEQ + TL - 1) / TL)  // 74 tiles along L

// LDS layout (bytes): A0 | B0 | Mu | A1 | B1 | red
#define OFF_A0 0
#define OFF_B0 32768
#define OFF_MU 65536
#define OFF_A1 98304
#define OFF_B1 131072
#define OFF_RED 163840
#define SMEM_BYTES (163840 + 1024)

// ws layout (bytes): energy partials | pad | W1 frags | W2 frags | Wq frags
#define WS_W1 16384
#define WS_W2 (WS_W1 + 64 * 1024)   // 64 frags * 1KB
#define WS_WQ (WS_W2 + 32 * 1024)   // 32 frags * 1KB

#if defined(__has_builtin) && __has_builtin(__builtin_amdgcn_s_wait_asynccnt)
#define WAIT_ASYNC() __builtin_amdgcn_s_wait_asynccnt(0)
#else
#define WAIT_ASYNC() asm volatile("s_wait_asynccnt 0x0" ::: "memory")
#endif

__device__ __forceinline__ float softplus_f(float x) {
  return fmaxf(x, 0.0f) + log1pf(expf(-fabsf(x)));
}

// Build a 16x32 bf16 A-fragment from two f32 LDS rows (concat along K at 64).
// Per-lane element map (ISA 7.12.2): e<8 -> K=kbase+8*hh+e ; e>=8 -> K=kbase+16+8*hh+(e-8)
__device__ __forceinline__ v16bf frag_a_f32(const float* __restrict__ rowA,
                                            const float* __restrict__ rowB,
                                            int kbase, int hh) {
  int K0 = kbase + 8 * hh;
  int K1 = kbase + 16 + 8 * hh;
  const float4* s0 = (const float4*)((K0 < 64) ? (rowA + K0) : (rowB + (K0 - 64)));
  const float4* s1 = (const float4*)((K1 < 64) ? (rowA + K1) : (rowB + (K1 - 64)));
  float4 a0 = s0[0], a1 = s0[1], b0 = s1[0], b1 = s1[1];
  v16bf o;
  o[0] = (__bf16)a0.x; o[1] = (__bf16)a0.y; o[2]  = (__bf16)a0.z; o[3]  = (__bf16)a0.w;
  o[4] = (__bf16)a1.x; o[5] = (__bf16)a1.y; o[6]  = (__bf16)a1.z; o[7]  = (__bf16)a1.w;
  o[8] = (__bf16)b0.x; o[9] = (__bf16)b0.y; o[10] = (__bf16)b0.z; o[11] = (__bf16)b0.w;
  o[12] = (__bf16)b1.x; o[13] = (__bf16)b1.y; o[14] = (__bf16)b1.z; o[15] = (__bf16)b1.w;
  return o;
}

// Build a 16x32 bf16 A-fragment from a bf16 LDS row (two b128 loads).
__device__ __forceinline__ v16bf frag_a_bf16(const __bf16* __restrict__ row,
                                             int kbase, int hh) {
  union { uint4 u[2]; v16bf v; } t;
  t.u[0] = *(const uint4*)(row + kbase + 8 * hh);
  t.u[1] = *(const uint4*)(row + kbase + 16 + 8 * hh);
  return t.v;
}

// Load a pre-swizzled 32x16 bf16 B-fragment (two global b128 loads).
__device__ __forceinline__ v16bf frag_w(const __bf16* __restrict__ base,
                                        int fidx, int lane) {
  union { uint4 u[2]; v16bf v; } t;
  const uint4* p = (const uint4*)(base + (size_t)fidx * 512 + lane * 16);
  t.u[0] = p[0]; t.u[1] = p[1];
  return t.v;
}

// Pre-swizzle f32 weights into bf16 WMMA B-fragment layout.
// B-frag element map: lane = n16 + 16*hh ; element e -> K = kbase + (e&7) + 8*hh + 16*(e>>3), N = ntile*16 + n16.
__global__ __launch_bounds__(256)
void prep_weights(const float* __restrict__ W1, const float* __restrict__ W2,
                  const float* __restrict__ Wq,
                  __bf16* __restrict__ o1, __bf16* __restrict__ o2,
                  __bf16* __restrict__ oq) {
  int i = blockIdx.x * 256 + threadIdx.x;
  int rem = i & 511, lane = rem >> 4, e = rem & 15;
  int n16 = lane & 15, hh = lane >> 4;
  if (i < 64 * 512) {  // W1 [128x256]: frag (nt 0..15, kt 0..3)
    int f = i >> 9, nt = f >> 2, kt = f & 3;
    int k = kt * 32 + (e & 7) + 8 * hh + 16 * (e >> 3);
    o1[i] = (__bf16)W1[k * 256 + nt * 16 + n16];
  }
  if (i < 32 * 512) {  // W2 [256x64]: frag (nt 0..3, kt 0..7)
    int f = i >> 9, nt = f >> 3, kt = f & 7;
    int k = kt * 32 + (e & 7) + 8 * hh + 16 * (e >> 3);
    o2[i] = (__bf16)W2[k * 64 + nt * 16 + n16];
  }
  if (i < 8 * 512) {   // Wq [64x64]: frag (nt 0..3, kt 0..1)
    int f = i >> 9, nt = f >> 1, kt = f & 1;
    int k = kt * 32 + (e & 7) + 8 * hh + 16 * (e >> 3);
    oq[i] = (__bf16)Wq[k * 64 + nt * 16 + n16];
  }
}

__global__ __launch_bounds__(256)
void fif_fused_kernel(const float* __restrict__ A, const float* __restrict__ Bin,
                      const float* __restrict__ Q,
                      const float* __restrict__ b1, const float* __restrict__ b2,
                      const float* __restrict__ bq,
                      const __bf16* __restrict__ wF1, const __bf16* __restrict__ wF2,
                      const __bf16* __restrict__ wFq,
                      float* __restrict__ outA, float* __restrict__ outB,
                      float* __restrict__ ePart) {
  extern __shared__ char smem[];
  float* sA0 = (float*)(smem + OFF_A0);
  float* sB0 = (float*)(smem + OFF_B0);
  float* sMu = (float*)(smem + OFF_MU);
  float* sA1 = (float*)(smem + OFF_A1);
  float* sB1 = (float*)(smem + OFF_B1);
  float* sRed = (float*)(smem + OFF_RED);

  const int b  = blockIdx.y;
  const int t  = blockIdx.x;
  const int l0 = t * TL;
  const int tid = threadIdx.x;

  // ---------- Phase A1: async bulk copy of tile+halo into LDS (circular wrap) ----------
  {
    const char* gA = (const char*)(A   + (size_t)b * LSEQ * DDIM);
    const char* gB = (const char*)(Bin + (size_t)b * LSEQ * DDIM);
    const char* gQ = (const char*)(Q   + (size_t)b * LSEQ * DDIM);
    const uint32_t ldsBase = (uint32_t)(uintptr_t)smem;  // low 32 bits = LDS offset
    const int NV = ROWS * 16;  // 16 x b128 per 64-col row
    for (int i = tid; i < NV; i += 256) {
      int r = i >> 4, j = i & 15;
      int l = (l0 - HALO + r + LSEQ) % LSEQ;
      size_t gOff = (((size_t)l * 16 + j) << 4);
      uint32_t dOff = (uint32_t)((r * 16 + j) << 4);
      const char* pa = gA + gOff;
      const char* pb = gB + gOff;
      const char* pq = gQ + gOff;
      uint32_t la = ldsBase + OFF_A0 + dOff;
      uint32_t lb = ldsBase + OFF_B0 + dOff;
      uint32_t lq = ldsBase + OFF_MU + dOff;
      asm volatile("global_load_async_to_lds_b128 %0, %1, off" :: "v"(la), "v"(pa) : "memory");
      asm volatile("global_load_async_to_lds_b128 %0, %1, off" :: "v"(lb), "v"(pb) : "memory");
      asm volatile("global_load_async_to_lds_b128 %0, %1, off" :: "v"(lq), "v"(pq) : "memory");
    }
    WAIT_ASYNC();
  }
  __syncthreads();

  // ---------- Phase A2: mu via bf16 WMMA, one 16-row M-tile per wave ----------
  {
    const int w    = tid >> 5;
    const int lane = tid & 31;
    const int m16  = lane & 15;
    const int hh   = lane >> 4;
    const int mrow0 = w * 16;
    __bf16* stage = (__bf16*)(smem + OFF_A1) + (size_t)w * (16 * 256);
    const float* rA = sA0 + (mrow0 + m16) * 64;
    const float* rB = sB0 + (mrow0 + m16) * 64;
    const float* rQ = sMu + (mrow0 + m16) * 64;

    // GEMM1: [16x128] x [128x256] -> bias+relu -> bf16 stage
    v16bf a1[4];
#pragma unroll
    for (int kt = 0; kt < 4; ++kt) a1[kt] = frag_a_f32(rA, rB, kt * 32, hh);

    for (int nt = 0; nt < 16; ++nt) {
      v8f c = {};
#pragma unroll
      for (int kt = 0; kt < 4; ++kt)
        c = __builtin_amdgcn_wmma_f32_16x16x32_bf16(false, a1[kt], false,
                                                    frag_w(wF1, nt * 4 + kt, lane),
                                                    (short)0, c, false, false);
      int col = nt * 16 + m16;
      float bias = b1[col];
#pragma unroll
      for (int vr = 0; vr < 8; ++vr) {
        int row = vr + 8 * hh;
        float x = c[vr] + bias;
        x = x > 0.0f ? x : 0.0f;
        stage[row * 256 + col] = (__bf16)x;
      }
    }

    // GEMM2: [16x256] x [256x64]
    v8f c2[4];
#pragma unroll
    for (int nt = 0; nt < 4; ++nt)
#pragma unroll
      for (int i = 0; i < 8; ++i) c2[nt][i] = 0.0f;
    for (int kt = 0; kt < 8; ++kt) {
      v16bf a2 = frag_a_bf16(stage + m16 * 256, kt * 32, hh);
#pragma unroll
      for (int nt = 0; nt < 4; ++nt)
        c2[nt] = __builtin_amdgcn_wmma_f32_16x16x32_bf16(false, a2, false,
                                                         frag_w(wF2, nt * 8 + kt, lane),
                                                         (short)0, c2[nt], false, false);
    }

    // Gate GEMM: q[16x64] x Wq[64x64]  (q staged in sMu)
    v16bf aq0 = frag_a_f32(rQ, rQ, 0, hh);
    v16bf aq1 = frag_a_f32(rQ, rQ, 32, hh);
    v8f c3[4];
#pragma unroll
    for (int nt = 0; nt < 4; ++nt)
#pragma unroll
      for (int i = 0; i < 8; ++i) c3[nt][i] = 0.0f;
#pragma unroll
    for (int nt = 0; nt < 4; ++nt) {
      c3[nt] = __builtin_amdgcn_wmma_f32_16x16x32_bf16(false, aq0, false,
                                                       frag_w(wFq, nt * 2 + 0, lane),
                                                       (short)0, c3[nt], false, false);
      c3[nt] = __builtin_amdgcn_wmma_f32_16x16x32_bf16(false, aq1, false,
                                                       frag_w(wFq, nt * 2 + 1, lane),
                                                       (short)0, c3[nt], false, false);
    }

    // mu = softplus(s2) * (1 + 0.5*tanh(gate)) -> overwrite sMu (wave-local rows)
#pragma unroll
    for (int nt = 0; nt < 4; ++nt) {
      int col = nt * 16 + m16;
      float bb2 = b2[col], bbq = bq[col];
#pragma unroll
      for (int vr = 0; vr < 8; ++vr) {
        int row = mrow0 + vr + 8 * hh;
        float sp = softplus_f(c2[nt][vr] + bb2);
        float g  = tanhf(c3[nt][vr] + bbq);
        sMu[row * 64 + col] = sp * (1.0f + 0.5f * g);
      }
    }
  }
  __syncthreads();

  // ---------- Phase B: K=6 stencil iterations, double-buffered LDS ----------
  {
    float* Ab[2] = { sA0, sA1 };
    float* Bb[2] = { sB0, sB1 };
    const int c = tid & 63;
    const int g = tid >> 6;  // 4 row strips
    int cur = 0;
    for (int k = 0; k < KITER; ++k) {
      float* aC = Ab[cur];     float* bC = Bb[cur];
      float* aN = Ab[cur ^ 1]; float* bN = Bb[cur ^ 1];
      const int lo = k + 1, hi = (ROWS - 1) - k;  // write [lo,hi)
      const int nrows = hi - lo;
      const int chunk = (nrows + 3) >> 2;
      int r0 = lo + g * chunk;
      int r1 = r0 + chunk; if (r1 > hi) r1 = hi;
      if (r0 < hi) {
        float uAp, uBp, uAc, uBc;
        {
          int r = r0 - 1;
          float a = aC[r * 64 + c], bb = bC[r * 64 + c], m = sMu[r * 64 + c];
          float s = ETA * m * (a - bb); uAp = a - s; uBp = bb + s;
        }
        {
          int r = r0;
          float a = aC[r * 64 + c], bb = bC[r * 64 + c], m = sMu[r * 64 + c];
          float s = ETA * m * (a - bb); uAc = a - s; uBc = bb + s;
        }
        for (int r = r0; r < r1; ++r) {
          int rr = r + 1;
          float a = aC[rr * 64 + c], bb = bC[rr * 64 + c], m = sMu[rr * 64 + c];
          float s = ETA * m * (a - bb);
          float uAn = a - s, uBn = bb + s;
          aN[r * 64 + c] = uAc - LAM * (2.0f * uAc - uAp - uAn);
          bN[r * 64 + c] = uBc - LAM * (2.0f * uBc - uBp - uBn);
          uAp = uAc; uBp = uBc; uAc = uAn; uBc = uBn;
        }
      }
      __syncthreads();
      cur ^= 1;
    }

    // ---------- Phase C: store results + deterministic energy partials ----------
    float* aF = Ab[cur];
    float* bF = Bb[cur];
    int tlEff = LSEQ - l0; if (tlEff > TL) tlEff = TL;
    const size_t base = (size_t)b * LSEQ * DDIM + (size_t)l0 * DDIM;
    float4* oA = (float4*)(outA + base);
    float4* oB = (float4*)(outB + base);
    float esum = 0.0f;
    const int NV = tlEff * 16;
    for (int i = tid; i < NV; i += 256) {
      int r = HALO + (i >> 4), j = i & 15;
      float4 va = ((float4*)aF)[r * 16 + j];
      float4 vb = ((float4*)bF)[r * 16 + j];
      float4 vm = ((float4*)sMu)[r * 16 + j];
      oA[i] = va;
      oB[i] = vb;
      float dx = va.x - vb.x, dy = va.y - vb.y, dz = va.z - vb.z, dw = va.w - vb.w;
      esum += 0.5f * (vm.x * dx * dx + vm.y * dy * dy + vm.z * dz * dz + vm.w * dw * dw);
    }
    sRed[tid] = esum;
    __syncthreads();
    for (int s = 128; s > 0; s >>= 1) {
      if (tid < s) sRed[tid] += sRed[tid + s];
      __syncthreads();
    }
    if (tid == 0) ePart[b * NT_L + t] = sRed[0];
  }
}

__global__ void fif_energy_finalize(const float* __restrict__ part,
                                    float* __restrict__ eOut) {
  int b = threadIdx.x;
  if (b < BSZ) {
    float s = 0.0f;
    for (int t = 0; t < NT_L; ++t) s += part[b * NT_L + t];
    eOut[b] = s;
  }
}

extern "C" void kernel_launch(void* const* d_in, const int* in_sizes, int n_in,
                              void* d_out, int out_size, void* d_ws, size_t ws_size,
                              hipStream_t stream) {
  const float* A   = (const float*)d_in[0];
  const float* Bin = (const float*)d_in[1];
  const float* Q   = (const float*)d_in[2];
  const float* W1  = (const float*)d_in[3];
  const float* b1  = (const float*)d_in[4];
  const float* W2  = (const float*)d_in[5];
  const float* b2  = (const float*)d_in[6];
  const float* Wq  = (const float*)d_in[7];
  const float* bq  = (const float*)d_in[8];

  float* out  = (float*)d_out;
  float* outA = out;
  float* outB = out + (size_t)BSZ * LSEQ * DDIM;
  float* eOut = outB + (size_t)BSZ * LSEQ * DDIM;

  char* ws = (char*)d_ws;
  float*  part = (float*)ws;                 // BSZ * NT_L partials
  __bf16* wF1  = (__bf16*)(ws + WS_W1);
  __bf16* wF2  = (__bf16*)(ws + WS_W2);
  __bf16* wFq  = (__bf16*)(ws + WS_WQ);

  prep_weights<<<128, 256, 0, stream>>>(W1, W2, Wq, wF1, wF2, wFq);

  dim3 grid(NT_L, BSZ);
  fif_fused_kernel<<<grid, 256, SMEM_BYTES, stream>>>(
      A, Bin, Q, b1, b2, bq, wF1, wF2, wFq, outA, outB, part);

  fif_energy_finalize<<<1, 32, 0, stream>>>(part, eOut);
}
